// AttentionLayer_88768384074430
// MI455X (gfx1250) — compile-verified, matrix operand order checked by reading, not run on hardware
//
#include <hip/hip_runtime.h>
#include <hip/hip_bf16.h>

// ---------------------------------------------------------------------------
// AttentionLayer forward for MI455X (gfx1250): bf16 WMMA GEMMs + windowed
// flash attention. f32 in / f32 out, bf16 tensor-core compute.
// ---------------------------------------------------------------------------

typedef __attribute__((ext_vector_type(16))) __bf16        v16bf;
typedef __attribute__((ext_vector_type(8)))  float         v8f;
typedef __attribute__((ext_vector_type(4)))  unsigned int  u32x4;
typedef __attribute__((ext_vector_type(4)))  float         f32x4;

#define HEADS   16
#define DIMS    2048
#define HDIM    128
#define SEQ     2048
#define BATCH   2
#define WINDOW  256
#define MTOT    (BATCH * SEQ)   // 4096 rows in every GEMM

union FragU {
    u32x4 q[2];   // q[0] -> VGPRs 0-3 (first 8 bf16), q[1] -> VGPRs 4-7
    v16bf bf;
};

__device__ __forceinline__ unsigned short f2bf(float x) {
    unsigned int u = __builtin_bit_cast(unsigned int, x);
    u += 0x7FFFu + ((u >> 16) & 1u);           // round-to-nearest-even
    return (unsigned short)(u >> 16);
}

// Load one 16-bit A/B-style fragment (16 bf16 per lane) from LDS.
// Per ISA 16-bit 16x32 layout: lane half selects kbase (0 or 8); VGPRs 0-3
// hold K = kb..kb+7 and VGPRs 4-7 hold K = kb+16..kb+23 (contiguous -> 2x b128).
__device__ __forceinline__ v16bf load_frag_lds(const unsigned short* base,
                                               int stride_us, int row, int kb) {
    FragU f;
    const unsigned short* p = base + row * stride_us + kb;
    f.q[0] = *(const u32x4*)(p);
    f.q[1] = *(const u32x4*)(p + 16);
    return f.bf;
}

// ---------------------------------------------------------------------------
// C[M,N] = A[M,K] @ B[K,N] + bias ; A is f32 or bf16 (template), B/bias f32.
// Block tile 256x128, BK=32. 256 threads = 8 waves arranged 4(M) x 2(N);
// each wave owns a 64x64 output = 4x4 grid of 16x16 WMMA tiles
// (16 WMMA per stage vs 16 ds_load_b128 -> 1.0 WMMA per fragment load).
// ---------------------------------------------------------------------------
template <bool ABF16>
__global__ __launch_bounds__(256)
void gemm_bias_wmma(const float* __restrict__ A32,
                    const unsigned short* __restrict__ A16,
                    const float* __restrict__ B,
                    const float* __restrict__ bias,
                    float* __restrict__ Cf,
                    unsigned short* __restrict__ Cb,
                    int M, int N, int K)
{
    __shared__ __align__(16) unsigned short sA[256][40];   // [m][k] bf16, pad 8
    __shared__ __align__(16) unsigned short sBt[128][40];  // [n][k] bf16 (B^T)

    const int tid   = threadIdx.x;
    const int lane  = tid & 31;
    const int wave  = tid >> 5;
    const int waveM = wave >> 1;          // 0..3
    const int waveN = wave & 1;           // 0..1
    const int m0    = blockIdx.y * 256;
    const int n0    = blockIdx.x * 128;
    const int hi    = (lane >> 4) & 1;
    const int l15   = lane & 15;
    const int kb    = hi ? 8 : 0;

    v8f zero = {};
    v8f acc[4][4];
    for (int i = 0; i < 4; ++i)
        for (int j = 0; j < 4; ++j) acc[i][j] = zero;

    for (int k0 = 0; k0 < K; k0 += 32) {
        __syncthreads();
        // ---- stage A tile (256x32) -> bf16 LDS --------------------------
        if constexpr (ABF16) {
            #pragma unroll
            for (int rep = 0; rep < 4; ++rep) {
                int idx = tid + rep * 256;            // 0..1023
                int r   = idx >> 2;                   // 0..255
                int c8  = (idx & 3) * 8;              // 0..24
                u32x4 v = *(const u32x4*)(A16 + (size_t)(m0 + r) * K + k0 + c8);
                *(u32x4*)&sA[r][c8] = v;
            }
            if (k0 + 32 < K)
                __builtin_prefetch(A16 + (size_t)(m0 + tid) * K + k0 + 32, 0, 0);
        } else {
            #pragma unroll
            for (int rep = 0; rep < 8; ++rep) {
                int idx = tid + rep * 256;            // 0..2047
                int r   = idx >> 3;                   // 0..255
                int c4  = (idx & 7) * 4;              // 0..28
                f32x4 v = *(const f32x4*)(A32 + (size_t)(m0 + r) * K + k0 + c4);
                unsigned short t4[4] = { f2bf(v.x), f2bf(v.y), f2bf(v.z), f2bf(v.w) };
                *(unsigned long long*)&sA[r][c4] = *(unsigned long long*)t4;
            }
            if (k0 + 32 < K)
                __builtin_prefetch(A32 + (size_t)(m0 + tid) * K + k0 + 32, 0, 0);
        }
        // ---- stage B tile (32x128), transposed -> sBt[n][k] -------------
        #pragma unroll
        for (int rep = 0; rep < 4; ++rep) {
            int idx = tid + rep * 256;                // 0..1023
            int kk  = idx >> 5;                       // 0..31
            int c4  = (idx & 31) * 4;                 // 0..124
            f32x4 v = *(const f32x4*)(B + (size_t)(k0 + kk) * N + n0 + c4);
            sBt[c4 + 0][kk] = f2bf(v.x);
            sBt[c4 + 1][kk] = f2bf(v.y);
            sBt[c4 + 2][kk] = f2bf(v.z);
            sBt[c4 + 3][kk] = f2bf(v.w);
        }
        if (k0 + 32 < K)
            __builtin_prefetch(B + (size_t)(k0 + 32 + (tid >> 5)) * N + n0 + (tid & 31) * 4, 0, 0);
        __syncthreads();

        v16bf af[4], bfm[4];
        #pragma unroll
        for (int i = 0; i < 4; ++i)
            af[i] = load_frag_lds(&sA[0][0], 40, waveM * 64 + 16 * i + l15, kb);
        #pragma unroll
        for (int j = 0; j < 4; ++j)
            bfm[j] = load_frag_lds(&sBt[0][0], 40, waveN * 64 + 16 * j + l15, kb);

        #pragma unroll
        for (int i = 0; i < 4; ++i)
            #pragma unroll
            for (int j = 0; j < 4; ++j)
                acc[i][j] = __builtin_amdgcn_wmma_f32_16x16x32_bf16(
                    false, af[i], false, bfm[j], (short)0, acc[i][j], false, false);
    }

    // ---- epilogue: + bias, store f32 and/or bf16 ------------------------
    for (int j = 0; j < 4; ++j) {
        int n   = n0 + waveN * 64 + 16 * j + l15;
        float bn = bias[n];
        for (int i = 0; i < 4; ++i) {
            int mb = m0 + waveM * 64 + 16 * i + 8 * hi;
            #pragma unroll
            for (int r = 0; r < 8; ++r) {
                float val = acc[i][j][r] + bn;
                size_t off = (size_t)(mb + r) * N + n;
                if (Cf) Cf[off] = val;
                if (Cb) Cb[off] = f2bf(val);
            }
        }
    }
}

// ---------------------------------------------------------------------------
// Windowed flash attention. Grid (S/128, H, B), 256 threads = 8 waves.
// Each wave owns 16 query rows; block stages shared 32-key K/V tiles.
// K tile staged with CDNA5 async global->LDS copies (ASYNCcnt).
// ---------------------------------------------------------------------------
__global__ __launch_bounds__(256)
void attn_wmma(const unsigned short* __restrict__ qb,
               const unsigned short* __restrict__ kbuf,
               const unsigned short* __restrict__ vbuf,
               unsigned short* __restrict__ ob)
{
    __shared__ __align__(16) unsigned short sK [32][136];  // [k][d] bf16
    __shared__ __align__(16) unsigned short sVt[128][40];  // [d][k] bf16 (V^T)
    __shared__ __align__(16) unsigned short sP [8][16][32];// per-wave P tile

    const int tid  = threadIdx.x;
    const int lane = tid & 31;
    const int wave = tid >> 5;            // 0..7
    const int l15  = lane & 15;
    const int hi   = (lane >> 4) & 1;
    const int kb   = hi ? 8 : 0;

    const int b    = blockIdx.z;
    const int h    = blockIdx.y;
    const int q0   = blockIdx.x * 128;
    const int qw   = q0 + wave * 16;
    const int col0 = h * HDIM;

    const float scale = 0.08838834764831844f;   // 1/sqrt(128)

    v8f zero = {};
    float m8[8], l8[8];
    v8f acc[8];
    #pragma unroll
    for (int r = 0; r < 8; ++r) { m8[r] = -3.0e38f; l8[r] = 0.0f; }
    #pragma unroll
    for (int j = 0; j < 8; ++j) acc[j] = zero;

    // --- load this wave's Q fragments (16 rows x 128, 4 K-chunks) --------
    v16bf qf[4];
    {
        const unsigned short* qrow =
            qb + ((size_t)(b * SEQ) + qw + l15) * DIMS + col0;
        #pragma unroll
        for (int c = 0; c < 4; ++c) {
            FragU f;
            f.q[0] = *(const u32x4*)(qrow + 32 * c + kb);
            f.q[1] = *(const u32x4*)(qrow + 32 * c + kb + 16);
            qf[c] = f.bf;
        }
    }

    int kstart = q0 - WINDOW; if (kstart < 0) kstart = 0;  // multiple of 32
    const int kend = q0 + 128;

    for (int kb0 = kstart; kb0 < kend; kb0 += 32) {
        __syncthreads();
        // ---- stage K rows (async global->LDS) + V transposed ----------
        #pragma unroll
        for (int rep = 0; rep < 2; ++rep) {
            int idx = tid + rep * 256;               // 0..511
            int r   = idx >> 4;                      // 0..31 key within tile
            int c8  = (idx & 15) * 8;                // 0..120
            const size_t g = ((size_t)(b * SEQ) + kb0 + r) * DIMS + col0 + c8;
            // K: raw bf16 16B chunks, no conversion -> async DMA into LDS
            unsigned long long ga = (unsigned long long)(const void*)(kbuf + g);
            unsigned           lo = (unsigned)(uintptr_t)(const void*)&sK[r][c8];
            asm volatile("global_load_async_to_lds_b128 %0, %1, off"
                         :: "v"(lo), "v"(ga) : "memory");
            // V: needs transpose -> manual scatter
            u32x4 vv = *(const u32x4*)(vbuf + g);
            const unsigned short* pv = (const unsigned short*)&vv;
            #pragma unroll
            for (int e = 0; e < 8; ++e) sVt[c8 + e][r] = pv[e];
        }
        asm volatile("s_wait_asynccnt 0" ::: "memory");
        __syncthreads();

        // wave-uniform skip (keeps EXEC all-1s around WMMA)
        if (kb0 > qw + 15 || kb0 + 31 < qw - WINDOW) continue;

        // ---- scores: two 16x16 tiles, Dh=128 -> 4 K-chunks each --------
        v8f s[2]; s[0] = zero; s[1] = zero;
        #pragma unroll
        for (int t2 = 0; t2 < 2; ++t2)
            #pragma unroll
            for (int c = 0; c < 4; ++c) {
                v16bf kf = load_frag_lds(&sK[0][0], 136, 16 * t2 + l15,
                                         32 * c + kb);
                s[t2] = __builtin_amdgcn_wmma_f32_16x16x32_bf16(
                    false, qf[c], false, kf, (short)0, s[t2], false, false);
            }

        // ---- mask + online softmax (rows live in 16-lane halves) -------
        #pragma unroll
        for (int r = 0; r < 8; ++r) {
            int qi  = qw + r + 8 * hi;
            int k0i = kb0 + l15;
            int k1i = kb0 + 16 + l15;
            bool ok0 = (k0i <= qi) && (qi - k0i <= WINDOW);
            bool ok1 = (k1i <= qi) && (qi - k1i <= WINDOW);
            float s0 = ok0 ? s[0][r] * scale : -3.0e38f;
            float s1 = ok1 ? s[1][r] * scale : -3.0e38f;
            float v = fmaxf(s0, s1);
            #pragma unroll
            for (int off = 8; off > 0; off >>= 1)
                v = fmaxf(v, __shfl_xor(v, off, 16));
            float mn    = fmaxf(m8[r], v);
            float alpha = __expf(m8[r] - mn);
            float p0 = ok0 ? __expf(s0 - mn) : 0.0f;   // explicit zero: no NaN
            float p1 = ok1 ? __expf(s1 - mn) : 0.0f;
            float rs = p0 + p1;
            #pragma unroll
            for (int off = 8; off > 0; off >>= 1)
                rs += __shfl_xor(rs, off, 16);
            l8[r] = l8[r] * alpha + rs;
            m8[r] = mn;
            #pragma unroll
            for (int j = 0; j < 8; ++j) acc[j][r] *= alpha;   // rescale O
            sP[wave][r + 8 * hi][l15]      = f2bf(p0);
            sP[wave][r + 8 * hi][16 + l15] = f2bf(p1);
        }
        // make wave's own LDS P stores visible before fragment reload
        asm volatile("s_wait_dscnt 0" ::: "memory");

        // ---- P(16x32) x V(32x128) ------------------------------------
        v16bf pf = load_frag_lds(&sP[wave][0][0], 32, l15, kb);
        #pragma unroll
        for (int j = 0; j < 8; ++j) {
            v16bf vf = load_frag_lds(&sVt[0][0], 40, 16 * j + l15, kb);
            acc[j] = __builtin_amdgcn_wmma_f32_16x16x32_bf16(
                false, pf, false, vf, (short)0, acc[j], false, false);
        }
    }

    // ---- normalize + store attn_out (bf16, merged-head layout) ---------
    const size_t obase = ((size_t)(b * SEQ) + qw) * DIMS + col0;
    #pragma unroll
    for (int r = 0; r < 8; ++r) {
        float inv = 1.0f / l8[r];
        int qrow = r + 8 * hi;
        #pragma unroll
        for (int j = 0; j < 8; ++j)
            ob[obase + (size_t)qrow * DIMS + 16 * j + l15] =
                f2bf(acc[j][r] * inv);
    }
}

// ---------------------------------------------------------------------------
extern "C" void kernel_launch(void* const* d_in, const int* in_sizes, int n_in,
                              void* d_out, int out_size, void* d_ws, size_t ws_size,
                              hipStream_t stream)
{
    const float* q  = (const float*)d_in[0];
    const float* k  = (const float*)d_in[1];
    const float* v  = (const float*)d_in[2];
    const float* Wq = (const float*)d_in[3];
    const float* bq = (const float*)d_in[4];
    const float* Wk = (const float*)d_in[5];
    const float* bk = (const float*)d_in[6];
    const float* Wv = (const float*)d_in[7];
    const float* bv = (const float*)d_in[8];
    const float* Wo = (const float*)d_in[9];
    const float* bo = (const float*)d_in[10];

    const size_t NEL = (size_t)MTOT * DIMS;      // 8,388,608 per tensor

    float* out = (float*)d_out;                  // tuple order: out, kp, vp
    float* kp  = out + NEL;
    float* vp  = kp + NEL;

    unsigned short* qpb = (unsigned short*)d_ws; // bf16 mirrors in workspace
    unsigned short* kpb = qpb + NEL;
    unsigned short* vpb = kpb + NEL;
    unsigned short* aob = vpb + NEL;             // attn_out bf16

    dim3 gg(DIMS / 128, MTOT / 256);             // (16, 16)
    dim3 bb(256);

    // Projections: qp (bf16 only), kp/vp (f32 outputs + bf16 mirrors)
    gemm_bias_wmma<false><<<gg, bb, 0, stream>>>(q, nullptr, Wq, bq, nullptr, qpb,
                                                 MTOT, DIMS, DIMS);
    gemm_bias_wmma<false><<<gg, bb, 0, stream>>>(k, nullptr, Wk, bk, kp, kpb,
                                                 MTOT, DIMS, DIMS);
    gemm_bias_wmma<false><<<gg, bb, 0, stream>>>(v, nullptr, Wv, bv, vp, vpb,
                                                 MTOT, DIMS, DIMS);

    // Windowed flash attention
    dim3 ga(SEQ / 128, HEADS, BATCH);
    attn_wmma<<<ga, dim3(256), 0, stream>>>(qpb, kpb, vpb, aob);

    // Output projection
    gemm_bias_wmma<true><<<gg, bb, 0, stream>>>(nullptr, aob, Wo, bo, out, nullptr,
                                                MTOT, DIMS, DIMS);
}
